// SimpleGRUDecoderWithAttention_48412871360744
// MI455X (gfx1250) — matrix operand order repaired and checked by reading, not compiled, not run on hardware
//
#include <hip/hip_runtime.h>
#include <math.h>

// ---------------------------------------------------------------------------
// SimpleGRUDecoderWithAttention, one step.  MI455X / gfx1250, wave32.
// fp32 weights streamed from HBM once (non-temporal), converted to bf16
// in-flight, all GEMMs on v_wmma_f32_16x16x32_bf16 with fp32 accumulators.
// GEMM: 256 thr = 8 waves, BM=128 x BN=128 x BK=32, each wave owns 32x64 of C
// (8 WMMA / K-step), software-pipelined global->reg->LDS staging.
// Attention is bandwidth-bound -> VALU + wave32 shuffle reductions; weights
// marked NT so encoder_hs (134 MB) stays resident in the 192 MB L2 for pass 2.
// ---------------------------------------------------------------------------

typedef __attribute__((ext_vector_type(16))) __bf16       v16bf;
typedef __attribute__((ext_vector_type(8)))  float        v8f;
typedef __attribute__((ext_vector_type(4)))  unsigned int u32x4;
typedef __attribute__((ext_vector_type(4)))  float        f32x4;

// Problem sizes (fixed by the reference)
constexpr int V   = 32000;
constexpr int E   = 512;
constexpr int H   = 1024;
constexpr int B   = 256;
constexpr int S   = 128;
constexpr int XW  = E + H;   // 1536
constexpr int G3H = 3 * H;   // 3072

// GEMM tiling
constexpr int BM  = 128;
constexpr int BN  = 128;
constexpr int BK  = 32;
constexpr int LDK = BK + 8;  // rows 80B: 16B-aligned vectors, 20-bank stride (conflict-free)

union FragBF { v16bf v; u32x4 q[2]; };

// ---------------------------------------------------------------------------
// C[M,N](fp32) = A_bf16[M,K] x (TRANSB ? W[N,K]^T : W[K,N]) (+ bias)
// N, K compile-time -> address math folds to immediate offsets.
// ---------------------------------------------------------------------------
template <bool TRANSB, bool HASBIAS, int N, int K>
__global__ __launch_bounds__(256)
void wmma_gemm_bf16(const __bf16* __restrict__ A, const float* __restrict__ W,
                    const float* __restrict__ bias, float* __restrict__ C)
{
  __shared__ alignas(16) __bf16 As[BM][LDK];
  __shared__ alignas(16) __bf16 Bs[BN][LDK];

  const int t    = threadIdx.x;
  const int lane = t & 31;
  const int wid  = t >> 5;
  const int wm   = wid >> 1;                 // 0..3 : 32-row slice
  const int wn   = wid & 1;                  // 0..1 : 64-col slice
  const int mBlk = blockIdx.y * BM;
  const int nBlk = blockIdx.x * BN;

  // staging coords (per thread)
  const int ar = t >> 1;                     // A row   0..127
  const int ac = (t & 1) * 16;               // A col   {0,16}
  const int bn = t >> 1;                     // B "n"   0..127
  const int bk = (t & 1) * 16;               // B k     {0,16}

  // per-lane fragment coords (ISA 7.12.2 bf16 layouts)
  const int aRowL = lane & 15;
  const int aKb   = (lane >> 4) * 8;         // A runs {aKb..+7, aKb+16..+23}
  const int bColL = lane & 15;
  const int bKb   = (lane >> 4) * 8;         // B runs {bKb..+7, bKb+16..+23}

  const v8f vzero = {0.f,0.f,0.f,0.f,0.f,0.f,0.f,0.f};
  v8f acc[2][4];
#pragma unroll
  for (int mo = 0; mo < 2; ++mo)
#pragma unroll
    for (int no = 0; no < 4; ++no) acc[mo][no] = vzero;

  // double-buffer registers (global -> reg -> LDS pipeline)
  u32x4 aReg0, aReg1;
  f32x4 wReg[4];
  float wRegS[16];

  auto loadG = [&](int k0) {
    const __bf16* ap = A + (size_t)(mBlk + ar) * K + k0 + ac;
    aReg0 = *reinterpret_cast<const u32x4*>(ap);
    aReg1 = *reinterpret_cast<const u32x4*>(ap + 8);
    if constexpr (TRANSB) {
      const float* wp = W + (size_t)(nBlk + bn) * K + k0 + bk;
#pragma unroll
      for (int j = 0; j < 4; ++j)   // streamed once: non-temporal (keep L2 for enc)
        wReg[j] = __builtin_nontemporal_load(
            reinterpret_cast<const f32x4*>(wp + 4 * j));
    } else {
#pragma unroll
      for (int j = 0; j < 16; ++j)
        wRegS[j] = W[(size_t)(k0 + bk + j) * N + nBlk + bn];
    }
  };

  auto storeLDS = [&]() {
    *reinterpret_cast<u32x4*>(&As[ar][ac])     = aReg0;
    *reinterpret_cast<u32x4*>(&As[ar][ac + 8]) = aReg1;
    if constexpr (TRANSB) {
#pragma unroll
      for (int j = 0; j < 4; ++j) {
        Bs[bn][bk + 4 * j + 0] = (__bf16)wReg[j].x;
        Bs[bn][bk + 4 * j + 1] = (__bf16)wReg[j].y;
        Bs[bn][bk + 4 * j + 2] = (__bf16)wReg[j].z;
        Bs[bn][bk + 4 * j + 3] = (__bf16)wReg[j].w;
      }
    } else {
#pragma unroll
      for (int j = 0; j < 16; ++j) Bs[bn][bk + j] = (__bf16)wRegS[j];
    }
  };

  constexpr int KT = K / BK;
  loadG(0);
  for (int kt = 0; kt < KT; ++kt) {
    storeLDS();
    __syncthreads();
    if (kt + 1 < KT) loadG((kt + 1) * BK);   // overlap next tile with WMMAs

    FragBF afr[2], bfr[4];
#pragma unroll
    for (int mo = 0; mo < 2; ++mo) {
      const int r = wm * 32 + mo * 16 + aRowL;
      afr[mo].q[0] = *reinterpret_cast<const u32x4*>(&As[r][aKb]);
      afr[mo].q[1] = *reinterpret_cast<const u32x4*>(&As[r][aKb + 16]);
    }
#pragma unroll
    for (int no = 0; no < 4; ++no) {
      const int c = wn * 64 + no * 16 + bColL;
      bfr[no].q[0] = *reinterpret_cast<const u32x4*>(&Bs[c][bKb]);
      bfr[no].q[1] = *reinterpret_cast<const u32x4*>(&Bs[c][bKb + 16]);
    }
#pragma unroll
    for (int mo = 0; mo < 2; ++mo)
#pragma unroll
      for (int no = 0; no < 4; ++no)
        acc[mo][no] = __builtin_amdgcn_wmma_f32_16x16x32_bf16(
            false, afr[mo].v, false, bfr[no].v, (short)0, acc[mo][no],
            false, false);
    __syncthreads();
  }

  // epilogue: C/D layout -> M = (lane>>4)*8 + r, N = lane&15
  const int cRow0 = (lane >> 4) * 8;
  const int cCol  = lane & 15;
#pragma unroll
  for (int mo = 0; mo < 2; ++mo) {
#pragma unroll
    for (int no = 0; no < 4; ++no) {
      const int gn = nBlk + wn * 64 + no * 16 + cCol;
      float bv = 0.0f;
      if constexpr (HASBIAS) bv = bias[gn];
      const size_t base = (size_t)(mBlk + wm * 32 + mo * 16 + cRow0) * N + gn;
#pragma unroll
      for (int r = 0; r < 8; ++r)
        C[base + (size_t)r * N] = acc[mo][no][r] + bv;
    }
  }
}

// ---------------------------------------------------------------------------
// prep: embedding gather + bf16 conversion of prev_h.  One thread per (b,h).
// ---------------------------------------------------------------------------
__global__ __launch_bounds__(256)
void prep_embed(const int* __restrict__ token, const float* __restrict__ prev_h,
                const float* __restrict__ emb_table,
                __bf16* __restrict__ xbf, __bf16* __restrict__ phbf)
{
  const int i = blockIdx.x * 256 + threadIdx.x;   // < B*H
  phbf[i] = (__bf16)prev_h[i];
  if (i < B * E) {
    const int b = i >> 9;            // / E
    const int e = i & (E - 1);
    xbf[(size_t)b * XW + e] = (__bf16)emb_table[(size_t)token[b] * E + e];
  }
}

// ---------------------------------------------------------------------------
// attention: one workgroup per batch element.
//   scores[s] = dot(q[b], enc[b,s,:])  (wave-parallel dot, shuffle reduce)
//   masked softmax over S=128 in LDS
//   context[h] = sum_s attn[s] * enc[b,s,h]   (2nd pass; enc resident in L2)
// context written as bf16 into x_bf[:, E:E+H].
// ---------------------------------------------------------------------------
__global__ __launch_bounds__(256)
void attn_softmax_ctx(const float* __restrict__ q, const float* __restrict__ enc,
                      const unsigned char* __restrict__ mask,   // jax bool = 1B
                      __bf16* __restrict__ xbf)
{
  __shared__ float qsh[H];
  __shared__ float sc[S];
  __shared__ float red[8];
  __shared__ float sstat;

  const int b    = blockIdx.x;
  const int t    = threadIdx.x;
  const int lane = t & 31;
  const int wid  = t >> 5;

  for (int i = t; i < H; i += 256) qsh[i] = q[(size_t)b * H + i];
  __syncthreads();

  // ---- pass 1: scores ----
  for (int s = wid; s < S; s += 8) {
    const float* er = enc + ((size_t)b * S + s) * H;
    float acc = 0.f;
    for (int k = lane; k < H; k += 32) acc += er[k] * qsh[k];
#pragma unroll
    for (int off = 16; off > 0; off >>= 1) acc += __shfl_xor(acc, off, 32);
    if (lane == 0) sc[s] = mask[(size_t)b * S + s] ? acc : -1e9f;
  }
  __syncthreads();

  // ---- softmax over 128 scores ----
  const float v = (t < S) ? sc[t] : -INFINITY;
  float m = v;
#pragma unroll
  for (int off = 16; off > 0; off >>= 1) m = fmaxf(m, __shfl_xor(m, off, 32));
  if (lane == 0) red[wid] = m;
  __syncthreads();
  if (t == 0) {
    float mm = red[0];
    for (int i = 1; i < 8; ++i) mm = fmaxf(mm, red[i]);
    sstat = mm;
  }
  __syncthreads();
  const float mx = sstat;
  const float p  = (t < S) ? __expf(v - mx) : 0.f;
  float ss = p;
#pragma unroll
  for (int off = 16; off > 0; off >>= 1) ss += __shfl_xor(ss, off, 32);
  __syncthreads();                       // everyone has read mx
  if (lane == 0) red[wid] = ss;
  __syncthreads();
  if (t == 0) {
    float s2 = 0.f;
    for (int i = 0; i < 8; ++i) s2 += red[i];
    sstat = 1.0f / s2;
  }
  __syncthreads();
  if (t < S) sc[t] = p * sstat;          // attention weights
  __syncthreads();

  // ---- pass 2: context ----
  for (int h0 = t; h0 < H; h0 += 256) {
    const float* eb = enc + (size_t)b * S * H + h0;
    float cacc = 0.f;
    for (int s = 0; s < S; ++s) cacc += sc[s] * eb[(size_t)s * H];
    xbf[(size_t)b * XW + E + h0] = (__bf16)cacc;
  }
}

// ---------------------------------------------------------------------------
// fused GRU gate math; writes h (fp32, 2nd output) and h_bf16 for out-GEMM
// ---------------------------------------------------------------------------
__global__ __launch_bounds__(256)
void gru_gates(const float* __restrict__ gi, const float* __restrict__ gh,
               const float* __restrict__ prev_h, float* __restrict__ h_out,
               __bf16* __restrict__ h_bf)
{
  const int i = blockIdx.x * 256 + threadIdx.x;   // < B*H
  const int b = i >> 10;
  const int h = i & (H - 1);
  const size_t g = (size_t)b * G3H;
  const float ir  = gi[g + h],          hr = gh[g + h];
  const float iz  = gi[g + H + h],      hz = gh[g + H + h];
  const float in_ = gi[g + 2 * H + h],  hn = gh[g + 2 * H + h];
  const float r = 1.f / (1.f + __expf(-(ir + hr)));
  const float z = 1.f / (1.f + __expf(-(iz + hz)));
  const float n = tanhf(in_ + r * hn);
  const float hv = (1.f - z) * n + z * prev_h[i];
  h_out[i] = hv;
  h_bf[i]  = (__bf16)hv;
}

// ---------------------------------------------------------------------------
extern "C" void kernel_launch(void* const* d_in, const int* in_sizes, int n_in,
                              void* d_out, int out_size, void* d_ws, size_t ws_size,
                              hipStream_t stream)
{
  (void)in_sizes; (void)n_in; (void)out_size; (void)ws_size;

  const int*           token  = (const int*)d_in[0];
  const float*         prev_h = (const float*)d_in[1];
  const float*         enc    = (const float*)d_in[2];
  const unsigned char* maskp  = (const unsigned char*)d_in[3];
  const float*         embt   = (const float*)d_in[4];
  const float*         attnW  = (const float*)d_in[5];
  const float*         W_ih   = (const float*)d_in[6];
  const float*         W_hh   = (const float*)d_in[7];
  const float*         b_ih   = (const float*)d_in[8];
  const float*         b_hh   = (const float*)d_in[9];
  const float*         W_out  = (const float*)d_in[10];
  const float*         b_out  = (const float*)d_in[11];

  float* out   = (float*)d_out;                 // [B,V]
  float* h_out = out + (size_t)B * V;           // [B,H] (2nd tuple output)

  // workspace layout (all offsets 16B-aligned), total ~8.75 MB
  char* ws = (char*)d_ws;
  __bf16* x_bf  = (__bf16*)(ws + 0);            // [B, XW]   786432 B
  __bf16* ph_bf = (__bf16*)(ws + 786432);       // [B, H]    524288 B
  __bf16* h_bf  = (__bf16*)(ws + 1310720);      // [B, H]    524288 B
  float*  qbuf  = (float*) (ws + 1835008);      // [B, H]   1048576 B
  float*  gi    = (float*) (ws + 2883584);      // [B, 3H]  3145728 B
  float*  gh    = (float*) (ws + 6029312);      // [B, 3H]  3145728 B

  const dim3 blk(256);

  // 1) embedding gather + prev_h -> bf16
  prep_embed<<<dim3((B * H) / 256), blk, 0, stream>>>(token, prev_h, embt, x_bf, ph_bf);

  // 2) q = prev_h @ attn_W   (weight [K,N], non-transposed path, no bias)
  wmma_gemm_bf16<false, false, H, H><<<dim3(H / BN, B / BM), blk, 0, stream>>>(
      ph_bf, attnW, nullptr, qbuf);

  // 3) attention scores + softmax + context (-> x_bf[:, E:])
  attn_softmax_ctx<<<dim3(B), blk, 0, stream>>>(qbuf, enc, maskp, x_bf);

  // 4) gi = x @ W_ih^T + b_ih
  wmma_gemm_bf16<true, true, G3H, XW><<<dim3(G3H / BN, B / BM), blk, 0, stream>>>(
      x_bf, W_ih, b_ih, gi);

  // 5) gh = prev_h @ W_hh^T + b_hh
  wmma_gemm_bf16<true, true, G3H, H><<<dim3(G3H / BN, B / BM), blk, 0, stream>>>(
      ph_bf, W_hh, b_hh, gh);

  // 6) GRU gates -> h (fp32 out) + h_bf16
  gru_gates<<<dim3((B * H) / 256), blk, 0, stream>>>(gi, gh, prev_h, h_out, h_bf);

  // 7) out = h @ W_out^T + b_out   (dominant GEMM: 256 x 32000 x 1024)
  wmma_gemm_bf16<true, true, V, H><<<dim3(V / BN, B / BM), blk, 0, stream>>>(
      h_bf, W_out, b_out, out);
}
